// MultiHeadNet_46557445488815
// MI455X (gfx1250) — matrix operand order repaired and checked by reading, not compiled, not run on hardware
//
#include <hip/hip_runtime.h>
#include <stdint.h>

// ---------------------------------------------------------------------------
// MultiHeadNet for MI455X (gfx1250, wave32, WMMA + async-to-LDS).
//  - BN folded into weights; weights pre-swizzled into WMMA B-fragment order.
//  - gemm1: barrier-free K-loop; x tile staged f32 into per-wave LDS via
//    global_load_async_to_lds_b32 (double buffered, ASYNCcnt-paced), converted
//    to f16 A-fragments at consume time; 7x v_wmma_f32_16x16x32_f16 per step.
//  - heads: two fused WMMA stages per head, exec-masked scatter of the
//    selected head's output.
// ---------------------------------------------------------------------------

typedef __attribute__((ext_vector_type(16))) _Float16 v16h;
typedef __attribute__((ext_vector_type(8)))  _Float16 v8h;
typedef __attribute__((ext_vector_type(8)))  float    v8f;

union AF16 { v16h v; v8h h2[2]; };

#define EPSV 1e-5f
#define IN_DIM 2051
#define NT1 7               // 112 = 7*16 output cols for HIDDEN=100

// workspace layout (bytes)
#define OFF_WNETF 0u          // 64*7*512 f16   = 458752 B
#define OFF_BNET  458752u     // 128 f32        = 512 B
#define OFF_W1F   459264u     // 8*4*4*512 f16  = 131072 B
#define OFF_B1    590336u     // 8*64 f32       = 2048 B
#define OFF_W2F   592384u     // 8*2*8*512 f16  = 131072 B
#define OFF_B2    723456u     // 8*128 f32      = 4096 B
#define OFF_H     786432u     // 65536*128 f16  = 16777216 B
#define OFF_HEAD  17563648u   // 65536 i32      = 262144 B

__device__ __forceinline__ v8f wmma_f16(v16h a, v16h b, v8f c) {
  return __builtin_amdgcn_wmma_f32_16x16x32_f16(false, a, false, b,
                                                (short)0, c, false, false);
}

// ---------------- prep kernels: fold BN, swizzle to B-fragment order --------
// B-fragment (32x16 f16, 8 VGPRs/lane): lane = n + ((k>>3)&1)*16, half = (k&7) + (k>>4)*8.

__global__ void prep_wnet(const float* __restrict__ Wn, const float* __restrict__ var0,
                          _Float16* __restrict__ Wf) {
  int gid = blockIdx.x * 256 + threadIdx.x;
  if (gid >= 2048 * 112) return;
  int k = gid / 112, j = gid - k * 112;
  float v = 0.f;
  if (j < 100) v = Wn[k * 100 + j] * rsqrtf(var0[k] + EPSV);
  int kt = k >> 5, kk = k & 31, nt = j >> 4, nn = j & 15;
  Wf[((size_t)(kt * NT1 + nt) * 32 + nn + ((kk >> 3) & 1) * 16) * 16 + (kk & 7) + (kk >> 4) * 8] =
      (_Float16)v;
}

__global__ void prep_bnet(const float* __restrict__ Wn, const float* __restrict__ m0,
                          const float* __restrict__ var0, const float* __restrict__ bn,
                          float* __restrict__ bnp) {
  int j = threadIdx.x;             // 0..127
  float r = 0.f;
  if (j < 100) {
    float s = 0.f;
    for (int k = 0; k < 2048; ++k)
      s += m0[k] * rsqrtf(var0[k] + EPSV) * Wn[k * 100 + j];
    r = bn[j] - s;
  }
  bnp[j] = r;
}

__global__ void prep_w1(const float* __restrict__ W1, const float* __restrict__ var1,
                        _Float16* __restrict__ W1f) {
  int gid = blockIdx.x * 256 + threadIdx.x;     // 8*128*64 = 65536
  int hh = gid >> 13, rem = gid & 8191;
  int d = rem >> 6, o = rem & 63;
  float v = 0.f;
  if (d < 100 && o < 50) v = W1[((size_t)hh * 100 + d) * 50 + o] * rsqrtf(var1[d] + EPSV);
  int kt = d >> 5, kk = d & 31, nt = o >> 4, nn = o & 15;
  W1f[(((size_t)(hh * 4 + kt) * 4 + nt) * 32 + nn + ((kk >> 3) & 1) * 16) * 16 +
      (kk & 7) + (kk >> 4) * 8] = (_Float16)v;
}

__global__ void prep_b1(const float* __restrict__ W1, const float* __restrict__ m1,
                        const float* __restrict__ var1, const float* __restrict__ b1,
                        float* __restrict__ b1p) {
  int gid = blockIdx.x * 256 + threadIdx.x;
  if (gid >= 8 * 64) return;
  int hh = gid >> 6, o = gid & 63;
  float r = 0.f;
  if (o < 50) {
    float s = 0.f;
    for (int d = 0; d < 100; ++d)
      s += m1[d] * rsqrtf(var1[d] + EPSV) * W1[((size_t)hh * 100 + d) * 50 + o];
    r = b1[hh * 50 + o] - s;
  }
  b1p[gid] = r;
}

__global__ void prep_w2(const float* __restrict__ W2, const float* __restrict__ var2,
                        _Float16* __restrict__ W2f) {
  int gid = blockIdx.x * 256 + threadIdx.x;     // 8*64*128 = 65536
  int hh = gid >> 13, rem = gid & 8191;
  int o = rem >> 7, d = rem & 127;
  float v = 0.f;
  if (o < 50) v = W2[((size_t)hh * 50 + o) * 128 + d] * rsqrtf(var2[hh * 50 + o] + EPSV);
  int kt = o >> 5, kk = o & 31, nt = d >> 4, nn = d & 15;
  W2f[(((size_t)(hh * 2 + kt) * 8 + nt) * 32 + nn + ((kk >> 3) & 1) * 16) * 16 +
      (kk & 7) + (kk >> 4) * 8] = (_Float16)v;
}

__global__ void prep_b2(const float* __restrict__ W2, const float* __restrict__ m2,
                        const float* __restrict__ var2, const float* __restrict__ b2,
                        float* __restrict__ b2p) {
  int gid = blockIdx.x * 256 + threadIdx.x;
  if (gid >= 8 * 128) return;
  int hh = gid >> 7, d = gid & 127;
  float s = 0.f;
  for (int o = 0; o < 50; ++o)
    s += m2[hh * 50 + o] * rsqrtf(var2[hh * 50 + o] + EPSV) *
         W2[((size_t)hh * 50 + o) * 128 + d];
  b2p[gid] = b2[hh * 128 + d] - s;
}

// ---------------- stage 1: h = relu(xn @ Wnet' + bnet'), head index ---------
// Block: 256 thr (8 waves), 128 rows. Wave w owns rows [w*16, w*16+16) for
// both staging (async DMA into its private LDS buffer) and its WMMA M-tile,
// so the K-loop needs no workgroup barriers.

__global__ __launch_bounds__(256) void gemm1_kernel(
    const float* __restrict__ x, const _Float16* __restrict__ Wf,
    const float* __restrict__ bnp, _Float16* __restrict__ hbuf,
    int* __restrict__ head) {
  __shared__ alignas(32) float    ldsX[8][2][512];  // per-wave dbl-buf 16x32 f32
  __shared__ alignas(32) _Float16 ldsE[8][2048];    // per-wave 16x128 epilogue

  const int tid = threadIdx.x;
  const int w = tid >> 5, lane = tid & 31;
  const int row0 = blockIdx.x * 128;
  const int nn = lane & 15, hi = lane >> 4;

  if (tid < 128) {
    const float* xr = x + (size_t)(row0 + tid) * IN_DIM;
    int idx = ((xr[0] > 0.5f) ? 4 : 0) | ((xr[1] > 0.5f) ? 2 : 0) |
              ((xr[2] > 0.5f) ? 1 : 0);
    head[row0 + tid] = idx;
  }

  v8f acc[NT1];
#pragma unroll
  for (int i = 0; i < NT1; ++i) acc[i] = {};

  // uniform global base (SGPR pair), per-lane byte offset (VGPR)
  const float* xbase = x + (size_t)row0 * IN_DIM;
  const uint32_t voff0 = (uint32_t)((w * 16 * IN_DIM + 3 + lane) * 4);
  const uint32_t lbase0 = (uint32_t)(uintptr_t)&ldsX[w][0][lane];
  const uint32_t lbase1 = (uint32_t)(uintptr_t)&ldsX[w][1][lane];

  // prologue: stage kt=0 into buffer 0
#pragma unroll
  for (int i = 0; i < 16; ++i) {
    uint32_t la = lbase0 + (uint32_t)(i * 128);
    uint32_t vo = voff0 + (uint32_t)(i * IN_DIM * 4);
    asm volatile("global_load_async_to_lds_b32 %0, %1, %2"
                 :: "v"(la), "v"(vo), "s"(xbase) : "memory");
  }

  // A-fragment consume offsets (ISA 16-bit A 16x32 layout):
  //   lane<16: K in {0..7, 16..23};  lane>=16: K in {8..15, 24..31}
  const int cbase = nn * 32 + hi * 8;

  for (int kt = 0; kt < 64; ++kt) {
    const int buf = kt & 1;
    if (kt < 63) {  // stage next tile into the other buffer
      uint32_t lb = buf ? lbase0 : lbase1;
      uint32_t vo0 = voff0 + (uint32_t)((kt + 1) * 128);
#pragma unroll
      for (int i = 0; i < 16; ++i) {
        uint32_t la = lb + (uint32_t)(i * 128);
        uint32_t vo = vo0 + (uint32_t)(i * IN_DIM * 4);
        asm volatile("global_load_async_to_lds_b32 %0, %1, %2"
                     :: "v"(la), "v"(vo), "s"(xbase) : "memory");
      }
      asm volatile("s_wait_asynccnt 16" ::: "memory");  // buffer `buf` ready
    } else {
      asm volatile("s_wait_asynccnt 0" ::: "memory");
    }

    const float* px = &ldsX[w][buf][cbase];
    v8f lo = *(const v8f*)px;           // K group at +0
    v8f hv = *(const v8f*)(px + 16);    // K group at +16
    AF16 a;
    a.h2[0] = __builtin_convertvector(lo, v8h);
    a.h2[1] = __builtin_convertvector(hv, v8h);

    const _Float16* wb = Wf + (size_t)kt * (NT1 * 512) + lane * 16;
#pragma unroll
    for (int nt = 0; nt < NT1; ++nt) {
      v16h b = *(const v16h*)(wb + nt * 512);
      acc[nt] = wmma_f16(a.v, b, acc[nt]);
    }
  }

  // epilogue: +bias, relu, f16, pad cols 112..127 with zeros, store h
#pragma unroll
  for (int nt = 0; nt < NT1; ++nt) {
    float bj = bnp[nt * 16 + nn];
#pragma unroll
    for (int r = 0; r < 8; ++r) {
      float zv = acc[nt][r] + bj;
      zv = zv > 0.f ? zv : 0.f;
      ldsE[w][(r + hi * 8) * 128 + nt * 16 + nn] = (_Float16)zv;
    }
  }
  if (lane < 16) {
#pragma unroll
    for (int c = 112; c < 128; ++c) ldsE[w][lane * 128 + c] = (_Float16)0.f;
  }
  __syncthreads();
#pragma unroll
  for (int i = 0; i < 4; ++i) {
    int m = (lane >> 3) + i * 4;
    int c0 = (lane & 7) * 16;
    v16h vv = *(const v16h*)&ldsE[w][m * 128 + c0];
    *(v16h*)(hbuf + (size_t)(row0 + w * 16 + m) * 128 + c0) = vv;
  }
}

// ---------------- stage 2+3: per-head MLP, exec-masked select ---------------
// Block: 16 rows, wave w computes head w end-to-end.

__global__ __launch_bounds__(256) void heads_kernel(
    const _Float16* __restrict__ hbuf, const int* __restrict__ head,
    const _Float16* __restrict__ W1f, const float* __restrict__ b1p,
    const _Float16* __restrict__ W2f, const float* __restrict__ b2p,
    float* __restrict__ out) {
  __shared__ alignas(32) _Float16 ldsZ[8][1024];  // per-wave 16x64 z3 tile

  const int tid = threadIdx.x, w = tid >> 5, lane = tid & 31;
  const int n0 = blockIdx.x * 16;
  const int nn = lane & 15, hi = lane >> 4;

  // stage 3: z3 = relu(h @ W1'[w] + b1'[w]),  K = 128 (padded)
  v8f acc3[4];
#pragma unroll
  for (int i = 0; i < 4; ++i) acc3[i] = {};
#pragma unroll
  for (int kt = 0; kt < 4; ++kt) {
    AF16 a;
    const _Float16* hp = hbuf + (size_t)(n0 + nn) * 128 + kt * 32 + hi * 8;
    a.h2[0] = *(const v8h*)hp;
    a.h2[1] = *(const v8h*)(hp + 16);
    const _Float16* wb = W1f + (size_t)(w * 4 + kt) * (4 * 512) + lane * 16;
#pragma unroll
    for (int nt = 0; nt < 4; ++nt) {
      v16h b = *(const v16h*)(wb + nt * 512);
      acc3[nt] = wmma_f16(a.v, b, acc3[nt]);
    }
  }
#pragma unroll
  for (int nt = 0; nt < 4; ++nt) {
    float bj = b1p[w * 64 + nt * 16 + nn];
#pragma unroll
    for (int r = 0; r < 8; ++r) {
      float zv = acc3[nt][r] + bj;
      zv = zv > 0.f ? zv : 0.f;
      ldsZ[w][(r + hi * 8) * 64 + nt * 16 + nn] = (_Float16)zv;
    }
  }
  __syncthreads();

  // stage 4: out = z3 @ W2'[w] + b2'[w],  K = 64 (padded)
  v8f acc4[8];
#pragma unroll
  for (int i = 0; i < 8; ++i) acc4[i] = {};
#pragma unroll
  for (int kt = 0; kt < 2; ++kt) {
    AF16 a;
    const _Float16* zp = &ldsZ[w][nn * 64 + kt * 32 + hi * 8];
    a.h2[0] = *(const v8h*)zp;
    a.h2[1] = *(const v8h*)(zp + 16);
    const _Float16* wb = W2f + (size_t)(w * 2 + kt) * (8 * 512) + lane * 16;
#pragma unroll
    for (int nt = 0; nt < 8; ++nt) {
      v16h b = *(const v16h*)(wb + nt * 512);
      acc4[nt] = wmma_f16(a.v, b, acc4[nt]);
    }
  }

  // exec-masked scatter: only the wave whose head matches the row writes
#pragma unroll
  for (int r = 0; r < 8; ++r) {
    int m = r + hi * 8;
    bool mine = (head[n0 + m] == w);
#pragma unroll
    for (int nt = 0; nt < 8; ++nt) {
      int n = nt * 16 + nn;
      float val = acc4[nt][r] + b2p[w * 128 + n];
      if (mine) out[(size_t)(n0 + m) * 128 + n] = val;
    }
  }
}

// ---------------------------------------------------------------------------

extern "C" void kernel_launch(void* const* d_in, const int* in_sizes, int n_in,
                              void* d_out, int out_size, void* d_ws, size_t ws_size,
                              hipStream_t stream) {
  const float* x  = (const float*)d_in[0];
  const float* m0 = (const float*)d_in[1];
  const float* v0 = (const float*)d_in[2];
  const float* Wn = (const float*)d_in[3];
  const float* bn = (const float*)d_in[4];
  const float* m1 = (const float*)d_in[5];
  const float* v1 = (const float*)d_in[6];
  const float* W1 = (const float*)d_in[7];
  const float* b1 = (const float*)d_in[8];
  const float* m2 = (const float*)d_in[9];
  const float* v2 = (const float*)d_in[10];
  const float* W2 = (const float*)d_in[11];
  const float* b2 = (const float*)d_in[12];
  float* out = (float*)d_out;
  char* ws = (char*)d_ws;

  _Float16* Wf  = (_Float16*)(ws + OFF_WNETF);
  float*    bnp = (float*)(ws + OFF_BNET);
  _Float16* W1f = (_Float16*)(ws + OFF_W1F);
  float*    b1p = (float*)(ws + OFF_B1);
  _Float16* W2f = (_Float16*)(ws + OFF_W2F);
  float*    b2p = (float*)(ws + OFF_B2);
  _Float16* hb  = (_Float16*)(ws + OFF_H);
  int*      hd  = (int*)(ws + OFF_HEAD);

  prep_wnet<<<896, 256, 0, stream>>>(Wn, v0, Wf);
  prep_bnet<<<1, 128, 0, stream>>>(Wn, m0, v0, bn, bnp);
  prep_w1<<<256, 256, 0, stream>>>(W1, v1, W1f);
  prep_b1<<<2, 256, 0, stream>>>(W1, m1, v1, b1, b1p);
  prep_w2<<<256, 256, 0, stream>>>(W2, v2, W2f);
  prep_b2<<<4, 256, 0, stream>>>(W2, m2, v2, b2, b2p);

  gemm1_kernel<<<512, 256, 0, stream>>>(x, Wf, bnp, hb, hd);
  heads_kernel<<<4096, 256, 0, stream>>>(hb, hd, W1f, b1p, W2f, b2p, out);
}
